// ExprQuantizer_25323127177636
// MI455X (gfx1250) — compile-verified, matrix operand order checked by reading, not run on hardware
//
#include <hip/hip_runtime.h>
#include <math.h>

#define HID   64
#define NLOG  9        // NUM_BINS - 1 logits
#define LEAKY 0.01f

typedef _Float16 v16h __attribute__((ext_vector_type(16)));
typedef float    v8f  __attribute__((ext_vector_type(8)));

// D = A x B + C with A = W2^T (16x32 f16, bins as rows, 9 live), B = H (32x16 f16,
// hidden as rows, 16 positions as cols), K=64 via two accumulating WMMAs.
// C is pre-loaded with b2 (bias add done by the matrix pipe).
// Two position-tiles per wave iteration -> 32 positions per loop step per wave.
__global__ __launch_bounds__(256) void expr_quant_wmma(
    const float* __restrict__ expr,   // total
    const float* __restrict__ W1,     // 64
    const float* __restrict__ b1,     // 64
    const float* __restrict__ W2,     // 64*9 row-major (k, bin)
    const float* __restrict__ b2,     // 9
    float* __restrict__ probs,        // total*10
    float* __restrict__ maskout,      // total
    int total)
{
    const int lane  = threadIdx.x & 31;
    const int halfW = lane >> 4;
    const int n16   = lane & 15;
    const int wid   = (blockIdx.x * blockDim.x + threadIdx.x) >> 5;
    const int nwav  = (gridDim.x * blockDim.x) >> 5;

    // K indices this lane owns inside one K=32 16-bit WMMA operand
    // (ISA 7.12.2: v0..v3 -> K = 8*half + {0..7}, v4..v7 -> K = 16 + 8*half + {0..7})
    int kk[16];
#pragma unroll
    for (int e = 0; e < 16; ++e)
        kk[e] = (e < 8) ? (halfW * 8 + e) : (16 + halfW * 8 + (e - 8));

    // ---- loop-invariant A operand: W2^T, zero-padded for bins 9..15 ----
    v16h a_lo, a_hi;
#pragma unroll
    for (int e = 0; e < 16; ++e) {
        float wlo = (n16 < NLOG) ? W2[kk[e] * NLOG + n16] : 0.0f;
        float whi = (n16 < NLOG) ? W2[(kk[e] + 32) * NLOG + n16] : 0.0f;
        a_lo[e] = (_Float16)wlo;
        a_hi[e] = (_Float16)whi;
    }
    // per-lane W1/b1 for the 32 hidden units this lane ever materializes
    float w1lo[16], b1lo[16], w1hi[16], b1hi[16];
#pragma unroll
    for (int e = 0; e < 16; ++e) {
        w1lo[e] = W1[kk[e]];      b1lo[e] = b1[kk[e]];
        w1hi[e] = W1[kk[e] + 32]; b1hi[e] = b1[kk[e] + 32];
    }
    // b2 folded into the WMMA accumulator init: lane's C[r] maps to bin M = r + 8*half
    v8f cinit;
#pragma unroll
    for (int r = 0; r < 8; ++r) {
        const int m = r + 8 * halfW;
        cinit[r] = (m < NLOG) ? b2[m] : 0.0f;
    }

    const int ngroups = total >> 5;           // 32 positions per wave-iteration
    for (int g = wid; g < ngroups; g += nwav) {
        const int base = g << 5;
        const float x0 = expr[base + n16];          // tile0, position n16
        const float x1 = expr[base + 16 + n16];     // tile1, position n16

        // ---- build B operands: h = LeakyReLU(x*W1 + b1) = max(h, 0.01*h) ----
        v16h bl0, bh0, bl1, bh1;
#pragma unroll
        for (int e = 0; e < 16; ++e) {
            float h;
            h = fmaf(x0, w1lo[e], b1lo[e]); h = fmaxf(h, LEAKY * h); bl0[e] = (_Float16)h;
            h = fmaf(x0, w1hi[e], b1hi[e]); h = fmaxf(h, LEAKY * h); bh0[e] = (_Float16)h;
            h = fmaf(x1, w1lo[e], b1lo[e]); h = fmaxf(h, LEAKY * h); bl1[e] = (_Float16)h;
            h = fmaf(x1, w1hi[e], b1hi[e]); h = fmaxf(h, LEAKY * h); bh1[e] = (_Float16)h;
        }

        v8f c0 = cinit;
        c0 = __builtin_amdgcn_wmma_f32_16x16x32_f16(false, a_lo, false, bl0, (short)0, c0, false, false);
        c0 = __builtin_amdgcn_wmma_f32_16x16x32_f16(false, a_hi, false, bh0, (short)0, c0, false, false);
        v8f c1 = cinit;
        c1 = __builtin_amdgcn_wmma_f32_16x16x32_f16(false, a_lo, false, bl1, (short)0, c1, false, false);
        c1 = __builtin_amdgcn_wmma_f32_16x16x32_f16(false, a_hi, false, bh1, (short)0, c1, false, false);

        // ---- epilogue: lanes 0..15 own tile0 positions, lanes 16..31 own tile1 ----
        float t8 = __shfl_xor(c0[0], 16);           // low half receives bin8 (+b2[8]) of tile0
        float sw[8];
#pragma unroll
        for (int r = 0; r < 8; ++r) sw[r] = __shfl_xor(c1[r], 16); // high half gets tile1 bins0..7
        const bool hi = (halfW != 0);

        float l[NLOG];
#pragma unroll
        for (int r = 0; r < 8; ++r) l[r] = hi ? sw[r] : c0[r];
        l[8] = hi ? c1[0] : t8;

        const float xv  = hi ? x1 : x0;
        const float msk = (xv != 0.0f) ? 1.0f : 0.0f;

        float mx = l[0];
#pragma unroll
        for (int k = 1; k < NLOG; ++k) mx = fmaxf(mx, l[k]);
        float p[NLOG], s = 0.f;
#pragma unroll
        for (int k = 0; k < NLOG; ++k) { p[k] = __expf(l[k] - mx); s += p[k]; }
        const float scale = msk * __frcp_rn(s);

        const int pos = base + lane;
        float2* o = (float2*)(probs + (size_t)pos * 10);   // 40B stride -> 8B aligned
        o[0] = make_float2(1.0f - msk, p[0] * scale);
        o[1] = make_float2(p[1] * scale, p[2] * scale);
        o[2] = make_float2(p[3] * scale, p[4] * scale);
        o[3] = make_float2(p[5] * scale, p[6] * scale);
        o[4] = make_float2(p[7] * scale, p[8] * scale);
        maskout[pos] = msk;
    }

    // ---- scalar f32 tail for total % 32 != 0 (dead for B*G = 2.56M) ----
    const int tailStart = ngroups << 5;
    for (int ppos = tailStart + blockIdx.x * blockDim.x + (int)threadIdx.x; ppos < total;
         ppos += gridDim.x * blockDim.x) {
        float x = expr[ppos];
        float msk = (x != 0.0f) ? 1.0f : 0.0f;
        float l[NLOG];
#pragma unroll
        for (int k = 0; k < NLOG; ++k) l[k] = b2[k];
        for (int j = 0; j < HID; ++j) {
            float h = fmaf(x, W1[j], b1[j]);
            h = fmaxf(h, LEAKY * h);
#pragma unroll
            for (int k = 0; k < NLOG; ++k) l[k] = fmaf(h, W2[j * NLOG + k], l[k]);
        }
        float mx = l[0];
        for (int k = 1; k < NLOG; ++k) mx = fmaxf(mx, l[k]);
        float s = 0.f, p[NLOG];
        for (int k = 0; k < NLOG; ++k) { p[k] = __expf(l[k] - mx); s += p[k]; }
        float scale = msk / s;
        float* o = probs + (size_t)ppos * 10;
        o[0] = 1.0f - msk;
        for (int k = 0; k < NLOG; ++k) o[1 + k] = p[k] * scale;
        maskout[ppos] = msk;
    }
}

extern "C" void kernel_launch(void* const* d_in, const int* in_sizes, int n_in,
                              void* d_out, int out_size, void* d_ws, size_t ws_size,
                              hipStream_t stream) {
    const float* expr = (const float*)d_in[0];
    const float* W1   = (const float*)d_in[1];
    const float* b1   = (const float*)d_in[2];
    const float* W2   = (const float*)d_in[3];
    const float* b2   = (const float*)d_in[4];

    const int total = in_sizes[0];                 // B*G
    float* probs    = (float*)d_out;               // total*10
    float* maskout  = probs + (size_t)total * 10;  // total

    const int threads = 256;                       // 8 wave32s per block
    int ngroups = total / 32;
    int blocks  = (ngroups + 7) / 8;
    if (blocks > 4096) blocks = 4096;              // grid-stride over 32-pos groups
    if (blocks < 1)    blocks = 1;
    expr_quant_wmma<<<blocks, threads, 0, stream>>>(expr, W1, b1, W2, b2,
                                                    probs, maskout, total);
}
